// FlashMultiHeadAttention_NSToken_45414984187906
// MI455X (gfx1250) — matrix operand order repaired and checked by reading, not compiled, not run on hardware
//
#include <hip/hip_runtime.h>
#include <hip/hip_bf16.h>

typedef _Float16 f16;
typedef __attribute__((ext_vector_type(16))) _Float16 v16h;
typedef __attribute__((ext_vector_type(8)))  _Float16 v8h;
typedef __attribute__((ext_vector_type(8)))  float    v8f;
typedef __attribute__((ext_vector_type(4)))  float    v4f;

#define B_   2
#define S_   2048
#define H_   1024
#define NH_  16
#define HD_  64
#define NTOK (B_*S_)

union AFrag { v16h v; v8h h[2]; };
static __device__ __forceinline__ v16h join8(v8h a, v8h b) {
    AFrag f; f.h[0] = a; f.h[1] = b; return f.v;
}

// -- CDNA5 async global->LDS copy (GV mode, 16B per lane, ASYNCcnt tracked) --
static __device__ __forceinline__ unsigned lds_off(const void* p) {
    // LDS aperture truncates flat addr[31:0] to the wave-relative LDS offset.
    return (unsigned)(size_t)p;
}
static __device__ __forceinline__ void async_load_b128(unsigned ldst,
                                                       const void* gsrc) {
    asm volatile("global_load_async_to_lds_b128 %0, %1, off"
                 :: "v"(ldst), "v"(gsrc) : "memory");
}
static __device__ __forceinline__ void wait_async0() {
    asm volatile("s_wait_asynccnt 0" ::: "memory");
}

// ---------------------------------------------------------------------------
// GEMM: Y = X(fp32)[NTOK,H] @ W(fp32)[H,H]^T + bias, output f16.
// TRANS=false -> out[b,h,s,d]  (token-major, head-dim contiguous)
// TRANS=true  -> out[b,h,d,s]  (transposed: V^T for the P*V B-fragments)
// Block: 64(M) x 64(N), 4 waves, k-step 32, v_wmma_f32_16x16x32_f16.
// ---------------------------------------------------------------------------
template<bool TRANS>
__global__ __launch_bounds__(128) void proj_kernel(
    const float* __restrict__ X, const float* __restrict__ W,
    const float* __restrict__ bias, f16* __restrict__ out)
{
    __shared__ f16 ldsX[64 * 40];   // pad 40 f16 / row (bank spread, 16B aligned)
    __shared__ f16 ldsW[64 * 40];
    const int tid = threadIdx.x;
    const int lane = tid & 31, wave = tid >> 5;
    const int half = lane >> 4, lane16 = lane & 15;
    const int m0 = blockIdx.x * 64;
    const int n0 = blockIdx.y * 64;

    v8f acc[4];
#pragma unroll
    for (int i = 0; i < 4; ++i)
#pragma unroll
        for (int r = 0; r < 8; ++r) acc[i][r] = 0.f;

    for (int k0 = 0; k0 < H_; k0 += 32) {
        __syncthreads();
#pragma unroll
        for (int p = 0; p < 4; ++p) {           // 64 rows x 8 float4 = 512 / 128
            int idx = tid + p * 128;
            int row = idx >> 3, c4 = (idx & 7) * 4;
            v4f xv = *(const v4f*)&X[(size_t)(m0 + row) * H_ + k0 + c4];
            v4f wv = *(const v4f*)&W[(size_t)(n0 + row) * H_ + k0 + c4];
            f16* px = &ldsX[row * 40 + c4];
            px[0] = (f16)xv.x; px[1] = (f16)xv.y; px[2] = (f16)xv.z; px[3] = (f16)xv.w;
            f16* pw = &ldsW[row * 40 + c4];
            pw[0] = (f16)wv.x; pw[1] = (f16)wv.y; pw[2] = (f16)wv.z; pw[3] = (f16)wv.w;
        }
        __syncthreads();
        // A fragment: row = lane&15, K chunks {half*8, 16+half*8}
        const int arow = wave * 16 + lane16;
        v8h a0 = *(const v8h*)&ldsX[arow * 40 + half * 8];
        v8h a1 = *(const v8h*)&ldsX[arow * 40 + 16 + half * 8];
        v16h Af = join8(a0, a1);
#pragma unroll
        for (int nt = 0; nt < 4; ++nt) {
            // B fragment: col n = lane&15, K chunks {half*8, 16+half*8} (row n of W)
            const int brow = nt * 16 + lane16;
            v8h b0 = *(const v8h*)&ldsW[brow * 40 + half * 8];
            v8h b1 = *(const v8h*)&ldsW[brow * 40 + 16 + half * 8];
            acc[nt] = __builtin_amdgcn_wmma_f32_16x16x32_f16(
                false, Af, false, join8(b0, b1), (short)0, acc[nt], false, false);
        }
    }

    // Epilogue: C layout -> vgpr r is row (m = r + 8*half), lane16 is column.
    const int t0 = m0 + wave * 16 + half * 8;     // 8 consecutive tokens t0..t0+7
    const int b = t0 >> 11, s0 = t0 & (S_ - 1);   // 64-blocks never straddle batch
#pragma unroll
    for (int nt = 0; nt < 4; ++nt) {
        int o = n0 + nt * 16 + lane16;
        float bv = bias[o];
        int h = o >> 6, d = o & (HD_ - 1);
        if (TRANS) {
            v8h vals;
#pragma unroll
            for (int r = 0; r < 8; ++r) vals[r] = (f16)(acc[nt][r] + bv);
            *(v8h*)&out[(((size_t)b * NH_ + h) * HD_ + d) * S_ + s0] = vals;
        } else {
#pragma unroll
            for (int r = 0; r < 8; ++r)
                out[(((size_t)b * NH_ + h) * S_ + s0 + r) * HD_ + d] =
                    (f16)(acc[nt][r] + bv);
        }
    }
}

// ---------------------------------------------------------------------------
// Flash attention: one block per (b*NH, 64 query rows); 4 waves x 16 rows.
// scores = (Q K^T + rel) * HD^-0.5, masked, online softmax, @V.
// K/V tiles staged with global_load_async_to_lds_b128 (ASYNCcnt).
// ---------------------------------------------------------------------------
__global__ __launch_bounds__(128) void attn_kernel(
    const f16* __restrict__ Q, const f16* __restrict__ K,
    const f16* __restrict__ Vt, const unsigned char* __restrict__ mask,
    const float* __restrict__ rel, f16* __restrict__ AO)
{
    __shared__ f16 ldsK[64 * 72];
    __shared__ f16 ldsV[64 * 72];
    __shared__ f16 ldsP[64 * 72];
    const int tid = threadIdx.x;
    const int lane = tid & 31, wave = tid >> 5;
    const int half = lane >> 4, lane16 = lane & 15;
    const int b = blockIdx.x >> 4, h = blockIdx.x & 15;
    const int q0 = blockIdx.y * 64 + wave * 16;   // wave's 16 query rows
    const size_t bh = (size_t)b * NH_ + h;

    // Q A-fragments straight from global (head-dim contiguous, 16B chunks)
    v16h aq[2];
    {
        const f16* qrow = Q + (bh * S_ + q0 + lane16) * HD_;
#pragma unroll
        for (int ks = 0; ks < 2; ++ks) {
            v8h c0 = *(const v8h*)&qrow[ks * 32 + half * 8];
            v8h c1 = *(const v8h*)&qrow[ks * 32 + 16 + half * 8];
            aq[ks] = join8(c0, c1);
        }
    }

    v8f oacc[4];
#pragma unroll
    for (int i = 0; i < 4; ++i)
#pragma unroll
        for (int r = 0; r < 8; ++r) oacc[i][r] = 0.f;
    float mrun[8], lrun[8];
#pragma unroll
    for (int r = 0; r < 8; ++r) { mrun[r] = -1e30f; lrun[r] = 0.f; }

    const float scale = 0.125f;                     // HD^-0.5
    const float L2E = 1.44269504088896340736f;

    // per-thread staging addresses (row/col fixed across kb)
    const int srow = tid >> 3, sc8 = (tid & 7) * 8;

    for (int kb = 0; kb < S_ / 64; ++kb) {
        __syncthreads();    // drains DScnt: prior-iter fragment reads complete
        {   // async-stage K tile [key][d] and V^T tile [d][key] (pad 72)
            const f16* kg = K + (bh * S_ + kb * 64) * HD_;   // 64*64 contiguous
            const f16* vg = Vt + bh * (size_t)HD_ * S_ + kb * 64;
#pragma unroll
            for (int p = 0; p < 4; ++p) {
                int row = srow + p * 16;
                async_load_b128(lds_off(&ldsK[row * 72 + sc8]),
                                &kg[row * 64 + sc8]);
                async_load_b128(lds_off(&ldsV[row * 72 + sc8]),
                                &vg[(size_t)row * S_ + sc8]);
            }
        }
        wait_async0();      // this wave's async tile writes landed in LDS
        __syncthreads();    // make all waves' tile writes visible

        // S = Q K^T : 4 n-tiles x 2 k-steps
        v8f sacc[4];
#pragma unroll
        for (int i = 0; i < 4; ++i)
#pragma unroll
            for (int r = 0; r < 8; ++r) sacc[i][r] = 0.f;
#pragma unroll
        for (int nt = 0; nt < 4; ++nt) {
#pragma unroll
            for (int ks = 0; ks < 2; ++ks) {
                int brow = nt * 16 + lane16;      // key index (col of B = K^T)
                v8h b0 = *(const v8h*)&ldsK[brow * 72 + ks * 32 + half * 8];
                v8h b1 = *(const v8h*)&ldsK[brow * 72 + ks * 32 + 16 + half * 8];
                sacc[nt] = __builtin_amdgcn_wmma_f32_16x16x32_f16(
                    false, aq[ks], false, join8(b0, b1), (short)0, sacc[nt], false, false);
            }
        }

        // bias + scale + mask (matches reference: (s + rel) * scale, then mask)
#pragma unroll
        for (int nt = 0; nt < 4; ++nt) {
#pragma unroll
            for (int r = 0; r < 8; ++r) {
                int q = q0 + r + half * 8;
                int kv = kb * 64 + nt * 16 + lane16;
                size_t idx = ((size_t)b * S_ + q) * S_ + kv;
                float sv = (sacc[nt][r] + rel[idx]) * scale;
                sacc[nt][r] = mask[idx] ? sv : -1e30f;
            }
        }

        // online softmax per row (16-lane row reductions via shfl_xor)
        float corr[8];
#pragma unroll
        for (int r = 0; r < 8; ++r) {
            float rm = fmaxf(fmaxf(sacc[0][r], sacc[1][r]),
                             fmaxf(sacc[2][r], sacc[3][r]));
#pragma unroll
            for (int m = 1; m < 16; m <<= 1) rm = fmaxf(rm, __shfl_xor(rm, m, 32));
            float mnew = fmaxf(mrun[r], rm);
            corr[r] = exp2f((mrun[r] - mnew) * L2E);
            float rs = 0.f;
#pragma unroll
            for (int nt = 0; nt < 4; ++nt) {
                float sv = sacc[nt][r];
                float p = (sv < -9e29f) ? 0.f : exp2f((sv - mnew) * L2E);
                sacc[nt][r] = p;
                rs += p;
            }
#pragma unroll
            for (int m = 1; m < 16; m <<= 1) rs += __shfl_xor(rs, m, 32);
            lrun[r] = lrun[r] * corr[r] + rs;
            mrun[r] = mnew;
        }
#pragma unroll
        for (int dt = 0; dt < 4; ++dt)
#pragma unroll
            for (int r = 0; r < 8; ++r) oacc[dt][r] *= corr[r];

        // P -> LDS (f16), re-read as A-fragments (wave-private rows, no barrier)
#pragma unroll
        for (int nt = 0; nt < 4; ++nt)
#pragma unroll
            for (int r = 0; r < 8; ++r)
                ldsP[(wave * 16 + r + half * 8) * 72 + nt * 16 + lane16] =
                    (f16)sacc[nt][r];

        // O += P V : A = P (16 x 64 keys), B = V^T rows are contiguous keys
#pragma unroll
        for (int ks = 0; ks < 2; ++ks) {
            int arow = wave * 16 + lane16;
            v8h a0 = *(const v8h*)&ldsP[arow * 72 + ks * 32 + half * 8];
            v8h a1 = *(const v8h*)&ldsP[arow * 72 + ks * 32 + 16 + half * 8];
            v16h Ap = join8(a0, a1);
#pragma unroll
            for (int dt = 0; dt < 4; ++dt) {
                int vrow = dt * 16 + lane16;      // d index (col of B)
                v8h b0 = *(const v8h*)&ldsV[vrow * 72 + ks * 32 + half * 8];
                v8h b1 = *(const v8h*)&ldsV[vrow * 72 + ks * 32 + 16 + half * 8];
                oacc[dt] = __builtin_amdgcn_wmma_f32_16x16x32_f16(
                    false, Ap, false, join8(b0, b1), (short)0, oacc[dt], false, false);
            }
        }
    }

    // normalize + store attn output f16 [B,S,H] (nan_to_num: fully-masked -> 0)
#pragma unroll
    for (int r = 0; r < 8; ++r) {
        float inv = lrun[r] > 0.f ? 1.f / lrun[r] : 0.f;
        int q = q0 + r + half * 8;
#pragma unroll
        for (int dt = 0; dt < 4; ++dt) {
            int o = h * HD_ + dt * 16 + lane16;
            AO[((size_t)b * S_ + q) * H_ + o] = (f16)(oacc[dt][r] * inv);
        }
    }
}

// ---------------------------------------------------------------------------
// Output projection: out(fp32) = AO(f16)[NTOK,H] @ Wo^T + bo
// X tile staged with async global->LDS (already f16); W tile converted in-VGPR.
// ---------------------------------------------------------------------------
__global__ __launch_bounds__(128) void oproj_kernel(
    const f16* __restrict__ X, const float* __restrict__ W,
    const float* __restrict__ bias, float* __restrict__ out)
{
    __shared__ f16 ldsX[64 * 40];
    __shared__ f16 ldsW[64 * 40];
    const int tid = threadIdx.x;
    const int lane = tid & 31, wave = tid >> 5;
    const int half = lane >> 4, lane16 = lane & 15;
    const int m0 = blockIdx.x * 64;
    const int n0 = blockIdx.y * 64;

    v8f acc[4];
#pragma unroll
    for (int i = 0; i < 4; ++i)
#pragma unroll
        for (int r = 0; r < 8; ++r) acc[i][r] = 0.f;

    for (int k0 = 0; k0 < H_; k0 += 32) {
        __syncthreads();
#pragma unroll
        for (int p = 0; p < 2; ++p) {            // X tile: 64x32 f16, async 16B/lane
            int idx = tid + p * 128;
            int row = idx >> 2, c8 = (idx & 3) * 8;
            async_load_b128(lds_off(&ldsX[row * 40 + c8]),
                            &X[(size_t)(m0 + row) * H_ + k0 + c8]);
        }
#pragma unroll
        for (int p = 0; p < 4; ++p) {            // W tile fp32 -> f16 (VGPR path)
            int idx = tid + p * 128;
            int row = idx >> 3, c4 = (idx & 7) * 4;
            v4f wv = *(const v4f*)&W[(size_t)(n0 + row) * H_ + k0 + c4];
            f16* pw = &ldsW[row * 40 + c4];
            pw[0] = (f16)wv.x; pw[1] = (f16)wv.y; pw[2] = (f16)wv.z; pw[3] = (f16)wv.w;
        }
        wait_async0();
        __syncthreads();
        const int arow = wave * 16 + lane16;
        v8h a0 = *(const v8h*)&ldsX[arow * 40 + half * 8];
        v8h a1 = *(const v8h*)&ldsX[arow * 40 + 16 + half * 8];
        v16h Af = join8(a0, a1);
#pragma unroll
        for (int nt = 0; nt < 4; ++nt) {
            const int brow = nt * 16 + lane16;
            v8h b0 = *(const v8h*)&ldsW[brow * 40 + half * 8];
            v8h b1 = *(const v8h*)&ldsW[brow * 40 + 16 + half * 8];
            acc[nt] = __builtin_amdgcn_wmma_f32_16x16x32_f16(
                false, Af, false, join8(b0, b1), (short)0, acc[nt], false, false);
        }
    }
    const int t0 = m0 + wave * 16 + half * 8;
#pragma unroll
    for (int nt = 0; nt < 4; ++nt) {
        int o = n0 + nt * 16 + lane16;
        float bv = bias[o];
#pragma unroll
        for (int r = 0; r < 8; ++r)
            out[(size_t)(t0 + r) * H_ + o] = acc[nt][r] + bv;
    }
}

// ---------------------------------------------------------------------------
extern "C" void kernel_launch(void* const* d_in, const int* in_sizes, int n_in,
                              void* d_out, int out_size, void* d_ws, size_t ws_size,
                              hipStream_t stream) {
    (void)in_sizes; (void)n_in; (void)out_size; (void)ws_size;
    const float* query = (const float*)d_in[0];
    const float* key   = (const float*)d_in[1];
    const float* value = (const float*)d_in[2];
    const unsigned char* mask = (const unsigned char*)d_in[3];  // bool array
    const float* rel = (const float*)d_in[4];
    const float* Wq = (const float*)d_in[5];  const float* bq = (const float*)d_in[6];
    const float* Wk = (const float*)d_in[7];  const float* bk = (const float*)d_in[8];
    const float* Wv = (const float*)d_in[9];  const float* bv = (const float*)d_in[10];
    const float* Wo = (const float*)d_in[11]; const float* bo = (const float*)d_in[12];

    const size_t perT = (size_t)B_ * NH_ * S_ * HD_;   // 4M f16 = 8 MB each
    f16* Qws  = (f16*)d_ws;
    f16* Kws  = Qws + perT;
    f16* Vws  = Kws + perT;
    f16* AOws = Vws + perT;                            // total 32 MB workspace

    dim3 pgrid(NTOK / 64, H_ / 64);
    proj_kernel<false><<<pgrid, 128, 0, stream>>>(query, Wq, bq, Qws);
    proj_kernel<false><<<pgrid, 128, 0, stream>>>(key,   Wk, bk, Kws);
    proj_kernel<true ><<<pgrid, 128, 0, stream>>>(value, Wv, bv, Vws);

    dim3 agrid(B_ * NH_, S_ / 64);
    attn_kernel<<<agrid, 128, 0, stream>>>(Qws, Kws, Vws, mask, rel, AOws);

    oproj_kernel<<<pgrid, 128, 0, stream>>>(AOws, Wo, bo, (float*)d_out);
}
